// DecoderWithAttention_73246372266587
// MI455X (gfx1250) — matrix operand order repaired and checked by reading, not compiled, not run on hardware
//
#include <hip/hip_runtime.h>
#include <math.h>

// ---------------------------------------------------------------------------
// Problem constants (from the reference)
// ---------------------------------------------------------------------------
#define BB   128
#define PP   196
#define ENCD 2048
#define AA   512
#define EE   512
#define DD   512
#define VV   10000
#define LL   22
#define TT   21   // decode steps = L-1

// ---------------------------------------------------------------------------
// WMMA fragment types (CDNA5 / gfx1250, wave32)
// ---------------------------------------------------------------------------
typedef __attribute__((ext_vector_type(16))) __bf16       v16bf;
typedef __attribute__((ext_vector_type(8)))  float        v8f;
typedef __attribute__((ext_vector_type(4)))  unsigned int v4u;

union Frag16 { v16bf v; v4u u4[2]; };
union AccF   { v8f   v; float f[8]; };

__device__ __forceinline__ float sigmoidf_(float x) {
    return 1.0f / (1.0f + __expf(-x));
}

// ---------------------------------------------------------------------------
// bf16 GEMM:  C[M,N] = A[M,K] * W[N,K]^T (+ bias), f32 accumulate via
// v_wmma_f32_16x16x32_bf16.
//   - wave tile: 32(M) x JN*16(N); block = 4 waves stacked in M -> 128 x JN*16
//   - JN is a compile-time template param (1..4); host splits N into full
//     64-wide blocks (JN=4) plus one tail block -> the inner K loop is
//     completely branch-free and the accumulators never cross a branch.
//   - K must be a multiple of 64 (true for all uses: 512 / 2048 / 2560);
//     K loop is double-buffered: loads for k+32 issue before WMMAs for k.
// flags bit0: accumulate into existing C.
// cap_len != nullptr: zero rows whose (cap_len[m]-1) <= t   (pred masking).
// ldc: row stride of C (strided write into preds[B,T,V]).
// ---------------------------------------------------------------------------
template<int JN>
__global__ __launch_bounds__(128)
void gemm_bf16_wmma(const __bf16* __restrict__ A, const __bf16* __restrict__ W,
                    const float* __restrict__ bias, float* __restrict__ C,
                    int M, int N, int K, int ldc, int flags,
                    const int* __restrict__ cap_len, int t, int nblk0)
{
    const int lane = threadIdx.x & 31;
    const int wave = threadIdx.x >> 5;
    const int m0 = blockIdx.y * 128 + wave * 32;          // wave-uniform
    const int n0 = (nblk0 + blockIdx.x) * 64;             // wave-uniform
    if (m0 >= M) return;                                  // uniform branch

    const int row = lane & 15;
    const int hi  = lane >> 4;

    const __bf16* arow0 = A + (size_t)(m0 + row) * K;     // rows m0   .. m0+15
    const __bf16* arow1 = arow0 + (size_t)16 * K;         // rows m0+16.. m0+31
    const __bf16* wrow  = W + (size_t)(n0 + row) * K;

    v8f acc0[JN] = {};
    v8f acc1[JN] = {};

    Frag16 a0A, a1A, bA[JN];     // set A (even k-steps)
    Frag16 a0B, a1B, bB[JN];     // set B (odd k-steps)

    auto loadA = [&](Frag16& f, const __bf16* base, int k) {
        // ISA 16-bit A layout: elems 0..7 = K[k+hi*8..], elems 8..15 = K[k+16+hi*8..]
        f.u4[0] = *(const v4u*)(base + k + hi * 8);
        f.u4[1] = *(const v4u*)(base + k + 16 + hi * 8);
    };
    auto loadB = [&](Frag16& f, int j, int k) {
        // B layout: lane holds col n = lane&15 (+j*16), elems 0..7 = K[k+hi*8..], elems 8..15 = K[k+16+hi*8..]
        const __bf16* p = wrow + (size_t)j * 16 * K + k + hi * 8;
        f.u4[0] = *(const v4u*)(p);
        f.u4[1] = *(const v4u*)(p + 16);
    };

    // prologue: fragments for k = 0
    loadA(a0A, arow0, 0);
    loadA(a1A, arow1, 0);
#pragma unroll
    for (int j = 0; j < JN; ++j) loadB(bA[j], j, 0);

    for (int k0 = 0; k0 < K; k0 += 64) {
        // prefetch k0+32 into set B
        loadA(a0B, arow0, k0 + 32);
        loadA(a1B, arow1, k0 + 32);
#pragma unroll
        for (int j = 0; j < JN; ++j) loadB(bB[j], j, k0 + 32);

        // math on set A
#pragma unroll
        for (int j = 0; j < JN; ++j)
            acc0[j] = __builtin_amdgcn_wmma_f32_16x16x32_bf16(
                false, a0A.v, false, bA[j].v, (short)0, acc0[j], false, false);
#pragma unroll
        for (int j = 0; j < JN; ++j)
            acc1[j] = __builtin_amdgcn_wmma_f32_16x16x32_bf16(
                false, a1A.v, false, bA[j].v, (short)0, acc1[j], false, false);

        // prefetch k0+64 into set A (uniform guard, accs untouched)
        if (k0 + 64 < K) {
            loadA(a0A, arow0, k0 + 64);
            loadA(a1A, arow1, k0 + 64);
#pragma unroll
            for (int j = 0; j < JN; ++j) loadB(bA[j], j, k0 + 64);
        }

        // math on set B
#pragma unroll
        for (int j = 0; j < JN; ++j)
            acc0[j] = __builtin_amdgcn_wmma_f32_16x16x32_bf16(
                false, a0B.v, false, bB[j].v, (short)0, acc0[j], false, false);
#pragma unroll
        for (int j = 0; j < JN; ++j)
            acc1[j] = __builtin_amdgcn_wmma_f32_16x16x32_bf16(
                false, a1B.v, false, bB[j].v, (short)0, acc1[j], false, false);
    }

    // store: C/D layout -> col = lane&15 (+j*16), rows = mbase + hi*8 + r
    auto storeTile = [&](v8f* acc, int mbase) {
#pragma unroll
        for (int j = 0; j < JN; ++j) {
            const int col = n0 + j * 16 + row;
            const float bval = bias ? bias[col] : 0.0f;
            AccF u; u.v = acc[j];
#pragma unroll
            for (int r = 0; r < 8; ++r) {
                const int m = mbase + hi * 8 + r;
                float val = u.f[r] + bval;
                if (cap_len && (cap_len[m] - 1) <= t) val = 0.0f;
                float* cp = C + (size_t)m * ldc + col;
                if (flags & 1) val += *cp;
                *cp = val;
            }
        }
    };
    storeTile(acc0, m0);
    storeTile(acc1, m0 + 16);
}

// ---------------------------------------------------------------------------
// fp32 -> bf16 elementwise convert
// ---------------------------------------------------------------------------
__global__ void f32_to_bf16_kernel(const float* __restrict__ in,
                                   __bf16* __restrict__ out, size_t n)
{
    size_t i = (size_t)blockIdx.x * blockDim.x + threadIdx.x;
    if (i < n) out[i] = (__bf16)in[i];
}

// mean over P of encoder_out -> bf16 [B,ENC]
__global__ void mean_enc_kernel(const float* __restrict__ enc,
                                __bf16* __restrict__ mean_bf)
{
    int i = blockIdx.x * blockDim.x + threadIdx.x;
    if (i >= BB * ENCD) return;
    int b = i >> 11, e = i & (ENCD - 1);
    const float* p = enc + (size_t)b * PP * ENCD + e;
    float s = 0.f;
    for (int k = 0; k < PP; ++k) s += p[(size_t)k * ENCD];
    mean_bf[i] = (__bf16)(s * (1.0f / (float)PP));
}

// embedding gather -> bf16 [B,L,E]
__global__ void embed_gather_kernel(const float* __restrict__ E_emb,
                                    const int* __restrict__ caps,
                                    __bf16* __restrict__ out)
{
    int i = blockIdx.x * blockDim.x + threadIdx.x;
    if (i >= BB * LL * EE) return;
    int e = i & (EE - 1);
    int bl = i >> 9;
    int tok = caps[bl];
    out[i] = (__bf16)E_emb[(size_t)tok * EE + e];
}

// ---------------------------------------------------------------------------
// Fused attention: e = relu(att1 + att2)·w_full, softmax over P, write masked
// alpha to output, awe = sum_p alpha*enc, build x = [emb_t | awe] in bf16.
// One block per batch row; 256 threads = 8 waves.
// ---------------------------------------------------------------------------
__global__ __launch_bounds__(256)
void attention_softmax_awe(const float* __restrict__ att1,
                           const float* __restrict__ att2,
                           const float* __restrict__ w_full,
                           const float* __restrict__ b_full,
                           const __bf16* __restrict__ enc_bf,
                           const __bf16* __restrict__ embs_bf,
                           __bf16* __restrict__ x_bf,
                           float* __restrict__ alphas_out,
                           const int* __restrict__ cap_len, int t)
{
    __shared__ float sAl[256];
    __shared__ float red[16];
    const int b = blockIdx.x;
    const int tid = threadIdx.x;
    const int wave = tid >> 5, lane = tid & 31;
    const bool active = (cap_len[b] - 1) > t;

    const float* a2 = att2 + (size_t)b * AA;
    for (int p = wave; p < PP; p += 8) {
        const float* a1 = att1 + ((size_t)b * PP + p) * AA;
        float s = 0.f;
        for (int k = lane; k < AA; k += 32) {
            float v = a1[k] + a2[k];
            v = v > 0.f ? v : 0.f;
            s += v * w_full[k];
        }
#pragma unroll
        for (int off = 16; off > 0; off >>= 1) s += __shfl_down(s, off, 32);
        if (lane == 0) sAl[p] = s + b_full[0];
    }
    __syncthreads();

    // softmax over 196 values
    float x  = (tid < PP) ? sAl[tid] : -3.0e38f;
    float mx = x;
#pragma unroll
    for (int off = 16; off > 0; off >>= 1) mx = fmaxf(mx, __shfl_xor(mx, off, 32));
    if (lane == 0) red[wave] = mx;
    __syncthreads();
    if (tid == 0) {
        float m = red[0];
        for (int i = 1; i < 8; ++i) m = fmaxf(m, red[i]);
        red[0] = m;
    }
    __syncthreads();
    const float gmax = red[0];
    float ev = (tid < PP) ? __expf(x - gmax) : 0.f;
    float ss = ev;
#pragma unroll
    for (int off = 16; off > 0; off >>= 1) ss += __shfl_xor(ss, off, 32);
    if (lane == 0) red[8 + wave] = ss;
    __syncthreads();
    if (tid == 0) {
        float s = 0.f;
        for (int i = 0; i < 8; ++i) s += red[8 + i];
        red[8] = 1.0f / s;
    }
    __syncthreads();
    const float inv = red[8];
    if (tid < PP) {
        float al = ev * inv;
        sAl[tid] = al;
        alphas_out[((size_t)b * TT + t) * PP + tid] = active ? al : 0.f;
    }
    __syncthreads();

    // awe[b,:] (ENC=2048) -> x_bf[b, E : E+ENC]
    const __bf16* eb = enc_bf + (size_t)b * PP * ENCD;
    for (int e = tid; e < ENCD; e += 256) {
        float s = 0.f;
        for (int p = 0; p < PP; ++p)
            s += sAl[p] * (float)eb[(size_t)p * ENCD + e];
        x_bf[(size_t)b * (EE + ENCD) + EE + e] = (__bf16)s;
    }
    // emb_t -> x_bf[b, 0:E]
    const __bf16* em = embs_bf + ((size_t)b * LL + t) * EE;
    for (int e = tid; e < EE; e += 256)
        x_bf[(size_t)b * (EE + ENCD) + e] = em[e];
}

// ---------------------------------------------------------------------------
// LSTM pointwise + ragged mask; writes fp32 h/c state and bf16 h for GEMMs
// ---------------------------------------------------------------------------
__global__ void lstm_cell_kernel(const float* __restrict__ gates,
                                 float* __restrict__ h, float* __restrict__ c,
                                 __bf16* __restrict__ h_bf,
                                 const int* __restrict__ cap_len, int t)
{
    int idx = blockIdx.x * blockDim.x + threadIdx.x;
    if (idx >= BB * DD) return;
    int b = idx >> 9;
    int d = idx & (DD - 1);
    const float* g = gates + (size_t)b * (4 * DD);
    float ig = sigmoidf_(g[d]);
    float fg = sigmoidf_(g[DD + d]);
    float gg = tanhf(g[2 * DD + d]);
    float og = sigmoidf_(g[3 * DD + d]);
    float cn = fg * c[idx] + ig * gg;
    float hn = og * tanhf(cn);
    bool active = (cap_len[b] - 1) > t;
    float h2 = active ? hn : h[idx];
    float c2 = active ? cn : c[idx];
    h[idx] = h2; c[idx] = c2; h_bf[idx] = (__bf16)h2;
}

// ---------------------------------------------------------------------------
// Host launcher
// ---------------------------------------------------------------------------
extern "C" void kernel_launch(void* const* d_in, const int* in_sizes, int n_in,
                              void* d_out, int out_size, void* d_ws, size_t ws_size,
                              hipStream_t stream)
{
    const float* enc       = (const float*)d_in[0];
    const int*   caps      = (const int*)  d_in[1];
    const int*   cap_len   = (const int*)  d_in[2];
    const float* E_emb     = (const float*)d_in[3];
    const float* W_enc_att = (const float*)d_in[4];
    const float* b_enc_att = (const float*)d_in[5];
    const float* W_dec_att = (const float*)d_in[6];
    const float* b_dec_att = (const float*)d_in[7];
    const float* w_full    = (const float*)d_in[8];
    const float* b_full    = (const float*)d_in[9];
    const float* W_init_h  = (const float*)d_in[10];
    const float* b_init_h  = (const float*)d_in[11];
    const float* W_init_c  = (const float*)d_in[12];
    const float* b_init_c  = (const float*)d_in[13];
    const float* W_ih      = (const float*)d_in[14];
    const float* b_ih      = (const float*)d_in[15];
    const float* W_hh      = (const float*)d_in[16];
    const float* b_hh      = (const float*)d_in[17];
    const float* W_fc      = (const float*)d_in[18];
    const float* b_fc      = (const float*)d_in[19];

    // Workspace bump allocator (256B aligned)
    char* base = (char*)d_ws;
    size_t off = 0;
    auto alloc = [&](size_t bytes) -> void* {
        void* p = base + off;
        off = (off + bytes + 255) & ~(size_t)255;
        return p;
    };

    __bf16* enc_bf  = (__bf16*)alloc((size_t)BB * PP * ENCD * 2);      // ~103 MB
    float*  att1    = (float*) alloc((size_t)BB * PP * AA * 4);        // ~51 MB
    __bf16* Wenc_bf = (__bf16*)alloc((size_t)AA * ENCD * 2);
    __bf16* Wdec_bf = (__bf16*)alloc((size_t)AA * DD * 2);
    __bf16* Wih_bf  = (__bf16*)alloc((size_t)4 * DD * (EE + ENCD) * 2);
    __bf16* Whh_bf  = (__bf16*)alloc((size_t)4 * DD * DD * 2);
    __bf16* Wfc_bf  = (__bf16*)alloc((size_t)VV * DD * 2);
    __bf16* Wh0_bf  = (__bf16*)alloc((size_t)DD * ENCD * 2);
    __bf16* Wc0_bf  = (__bf16*)alloc((size_t)DD * ENCD * 2);
    __bf16* mean_bf = (__bf16*)alloc((size_t)BB * ENCD * 2);
    __bf16* embs_bf = (__bf16*)alloc((size_t)BB * LL * EE * 2);
    float*  h_f     = (float*) alloc((size_t)BB * DD * 4);
    float*  c_f     = (float*) alloc((size_t)BB * DD * 4);
    __bf16* h_bf    = (__bf16*)alloc((size_t)BB * DD * 2);
    float*  att2    = (float*) alloc((size_t)BB * AA * 4);
    __bf16* x_bf    = (__bf16*)alloc((size_t)BB * (EE + ENCD) * 2);
    float*  gates   = (float*) alloc((size_t)BB * 4 * DD * 4);
    (void)ws_size; (void)in_sizes; (void)n_in; (void)out_size;

    auto cvt = [&](const float* in, __bf16* out, size_t n) {
        f32_to_bf16_kernel<<<(unsigned)((n + 255) / 256), 256, 0, stream>>>(in, out, n);
    };
    // GEMM dispatch: full 64-wide blocks with JN=4, optional 16/32/48-wide tail
    auto gemm = [&](const __bf16* Amat, const __bf16* Wmat, const float* bias,
                    float* C, int M, int N, int K, int ldc, int flags,
                    const int* mask, int t) {
        const int nfull = N / 64;
        const int gy = (M + 127) / 128;
        if (nfull > 0) {
            gemm_bf16_wmma<4><<<dim3(nfull, gy), 128, 0, stream>>>(
                Amat, Wmat, bias, C, M, N, K, ldc, flags, mask, t, 0);
        }
        const int rem = (N - nfull * 64) / 16;
        if (rem == 1)
            gemm_bf16_wmma<1><<<dim3(1, gy), 128, 0, stream>>>(
                Amat, Wmat, bias, C, M, N, K, ldc, flags, mask, t, nfull);
        else if (rem == 2)
            gemm_bf16_wmma<2><<<dim3(1, gy), 128, 0, stream>>>(
                Amat, Wmat, bias, C, M, N, K, ldc, flags, mask, t, nfull);
        else if (rem == 3)
            gemm_bf16_wmma<3><<<dim3(1, gy), 128, 0, stream>>>(
                Amat, Wmat, bias, C, M, N, K, ldc, flags, mask, t, nfull);
    };

    // -------- one-time precompute --------
    cvt(enc,       enc_bf,  (size_t)BB * PP * ENCD);       // bf16 copy fits in L2
    cvt(W_enc_att, Wenc_bf, (size_t)AA * ENCD);
    cvt(W_dec_att, Wdec_bf, (size_t)AA * DD);
    cvt(W_ih,      Wih_bf,  (size_t)4 * DD * (EE + ENCD));
    cvt(W_hh,      Whh_bf,  (size_t)4 * DD * DD);
    cvt(W_fc,      Wfc_bf,  (size_t)VV * DD);
    cvt(W_init_h,  Wh0_bf,  (size_t)DD * ENCD);
    cvt(W_init_c,  Wc0_bf,  (size_t)DD * ENCD);

    mean_enc_kernel<<<(BB * ENCD + 255) / 256, 256, 0, stream>>>(enc, mean_bf);
    embed_gather_kernel<<<(BB * LL * EE + 255) / 256, 256, 0, stream>>>(E_emb, caps, embs_bf);

    // h0 / c0
    gemm(mean_bf, Wh0_bf, b_init_h, h_f, BB, DD, ENCD, DD, 0, nullptr, 0);
    gemm(mean_bf, Wc0_bf, b_init_c, c_f, BB, DD, ENCD, DD, 0, nullptr, 0);
    cvt(h_f, h_bf, (size_t)BB * DD);

    // att1 = encoder_out @ W_enc_att^T + b  -> [B*P, A] fp32 (hoisted)
    gemm(enc_bf, Wenc_bf, b_enc_att, att1, BB * PP, AA, ENCD, AA, 0, nullptr, 0);

    float* preds  = (float*)d_out;                    // [B,T,V]
    float* alphas = preds + (size_t)BB * TT * VV;     // [B,T,P]

    // -------- 21 sequential decode steps --------
    for (int t = 0; t < TT; ++t) {
        // att2 = h @ W_dec_att^T + b  -> [B,A]
        gemm(h_bf, Wdec_bf, b_dec_att, att2, BB, AA, DD, AA, 0, nullptr, 0);

        attention_softmax_awe<<<BB, 256, 0, stream>>>(
            att1, att2, w_full, b_full, enc_bf, embs_bf, x_bf, alphas, cap_len, t);

        // gates = x @ W_ih^T + b_ih  then  += h @ W_hh^T + b_hh
        gemm(x_bf, Wih_bf, b_ih, gates, BB, 4 * DD, EE + ENCD, 4 * DD, 0, nullptr, 0);
        gemm(h_bf, Whh_bf, b_hh, gates, BB, 4 * DD, DD,        4 * DD, 1, nullptr, 0);

        lstm_cell_kernel<<<(BB * DD + 255) / 256, 256, 0, stream>>>(
            gates, h_f, c_f, h_bf, cap_len, t);

        // pred = mask(h_new @ W_fc^T + b_fc) strided into preds[:, t, :]
        gemm(h_bf, Wfc_bf, b_fc, preds + (size_t)t * VV, BB, VV, DD,
             TT * VV, 0, cap_len, t);
    }
}
